// TransformerBlock_56951266345562
// MI455X (gfx1250) — matrix operand order repaired and checked by reading, not compiled, not run on hardware
//
#include <hip/hip_runtime.h>
#include <cstdint>
#include <cstddef>

#define B_  2
#define S_  2048
#define D_  1024
#define H_  16
#define KH_ 4
#define HD_ 64
#define E_  8
#define FF_ 2048
#define T_  (B_*S_)

typedef __attribute__((ext_vector_type(16))) __bf16          bf16x16;
typedef __attribute__((ext_vector_type(4)))  unsigned short  u16x4;
typedef __attribute__((ext_vector_type(8)))  unsigned short  u16x8;
typedef __attribute__((ext_vector_type(16))) unsigned short  u16x16;
typedef __attribute__((ext_vector_type(8)))  float           f32x8;
typedef __attribute__((ext_vector_type(4)))  int             v4i;

// CDNA5 async global->LDS staging (bypasses VGPRs), guarded for toolchains
// that do not declare the gfx1250 builtins. Param types per hipcc diagnostic:
// (v4i addrspace(1)*, v4i addrspace(3)*, imm offset, imm cpol).
#if defined(__has_builtin)
#if __has_builtin(__builtin_amdgcn_global_load_async_to_lds_b128) && \
    __has_builtin(__builtin_amdgcn_s_wait_asynccnt)
#define HAVE_ASYNC_LDS 1
#endif
#endif
#ifndef HAVE_ASYNC_LDS
#define HAVE_ASYNC_LDS 0
#endif

#if HAVE_ASYNC_LDS
typedef __attribute__((address_space(1))) v4i* gv4i_p;
typedef __attribute__((address_space(3))) v4i* lv4i_p;
#define ASYNC_CP_B128(gp, lp)                                                   \
  __builtin_amdgcn_global_load_async_to_lds_b128((gv4i_p)(gp), (lv4i_p)(lp), 0, 0)
#define ASYNC_WAIT() __builtin_amdgcn_s_wait_asynccnt(0)
#else
#define ASYNC_CP_B128(gp, lp) (*(u16x8*)(lp) = *(const u16x8*)(gp))
#define ASYNC_WAIT() ((void)0)
#endif

static __device__ inline unsigned short f2bf(float f) {
  union { float f; unsigned u; } x; x.f = f;
  unsigned r = x.u + 0x7FFFu + ((x.u >> 16) & 1u);
  return (unsigned short)(r >> 16);
}
static __device__ inline float bf2f(unsigned short h) {
  union { unsigned u; float f; } x; x.u = ((unsigned)h) << 16;
  return x.f;
}
// Build a 16-element bf16 fragment from two contiguous 8-element LDS chunks.
static __device__ inline bf16x16 frag2(const unsigned short* p0, const unsigned short* p1) {
  u16x8 a = *(const u16x8*)p0;
  u16x8 b = *(const u16x8*)p1;
  u16x16 r;
#pragma unroll
  for (int i = 0; i < 8; i++) { r[i] = a[i]; r[i + 8] = b[i]; }
  return __builtin_bit_cast(bf16x16, r);
}
static __device__ inline f32x8 wmma_bf16(bf16x16 a, bf16x16 b, f32x8 c) {
  return __builtin_amdgcn_wmma_f32_16x16x32_bf16(false, a, false, b, (short)0, c, false, false);
}
// Coalesced f32->bf16 weight tile staging: lanes walk N (float4 loads),
// waves walk K, 4 K-adjacent bf16 packed into one 8B LDS store.
// Tile: 128(N) x 32(K), LDS layout [n][k] (B^T), 256 threads.
static __device__ inline void stage_w(const float* __restrict__ W, int N,
                                      int k0, int n0, unsigned short* lB,
                                      int lane, int wave) {
  int n4 = lane * 4;
  int kq = wave * 4;
  float4 wrow[4];
#pragma unroll
  for (int q = 0; q < 4; q++)
    wrow[q] = *(const float4*)&W[(size_t)(k0 + kq + q) * N + n0 + n4];
#pragma unroll
  for (int i = 0; i < 4; i++) {
    u16x4 pk;
#pragma unroll
    for (int q = 0; q < 4; q++) pk[q] = f2bf(((const float*)&wrow[q])[i]);
    *(u16x4*)&lB[(n4 + i) * 32 + kq] = pk;
  }
}

// ---------------------------------------------------------------- utilities
__global__ void k_zero(int* cnt, float* probsum) {
  int t = threadIdx.x;
  if (t < E_) { cnt[t] = 0; probsum[t] = 0.f; }
}

__global__ __launch_bounds__(256) void k_rmsnorm(const float* __restrict__ X,
                                                 const float* __restrict__ W,
                                                 unsigned short* __restrict__ Out) {
  int row = blockIdx.x;
  const float* x = X + (size_t)row * D_;
  int tid = threadIdx.x, lane = tid & 31, wave = tid >> 5;
  float s = 0.f;
  for (int d = tid; d < D_; d += 256) { float v = x[d]; s += v * v; }
#pragma unroll
  for (int dd = 1; dd < 32; dd <<= 1) s += __shfl_xor(s, dd, 32);
  __shared__ float red[8];
  __shared__ float rsv;
  if (lane == 0) red[wave] = s;
  __syncthreads();
  if (tid == 0) {
    float t = 0.f;
#pragma unroll
    for (int i = 0; i < 8; i++) t += red[i];
    rsv = rsqrtf(t / (float)D_ + 1e-6f);
  }
  __syncthreads();
  float rs = rsv;
  for (int d = tid; d < D_; d += 256)
    Out[(size_t)row * D_ + d] = f2bf(x[d] * rs * W[d]);
}

// ------------------------------------------------------- generic bf16 GEMM
// C[M x N] = bf16(A[M x K]) * bf16(W[K x N]) (+ resid), f32 out.
// block tile 64(M) x 128(N), 8 waves, each wave 32x32 via 2x2 WMMA frags.
__global__ __launch_bounds__(256) void k_gemm(const unsigned short* __restrict__ A,
                                              const float* __restrict__ W,
                                              float* __restrict__ C,
                                              const float* __restrict__ resid,
                                              int N, int K) {
  int m0 = blockIdx.x * 64;
  int n0 = blockIdx.y * 128;
  __shared__ __attribute__((aligned(16))) unsigned short lA[64 * 32];
  __shared__ __attribute__((aligned(16))) unsigned short lB[128 * 32];
  int tid = threadIdx.x, lane = tid & 31, wave = tid >> 5;
  int wm = (wave >> 2) * 32, wn = (wave & 3) * 32;
  int m = lane & 15, hi = lane >> 4;
  f32x8 acc[2][2];
#pragma unroll
  for (int i = 0; i < 2; i++)
#pragma unroll
    for (int j = 0; j < 2; j++) acc[i][j] = (f32x8)0.0f;

  for (int k0 = 0; k0 < K; k0 += 32) {
    {
      int r = tid >> 2, c = (tid & 3) * 8;
      ASYNC_CP_B128(&A[(size_t)(m0 + r) * K + k0 + c], &lA[r * 32 + c]);
    }
    stage_w(W, N, k0, n0, lB, lane, wave);
    if (k0 + 32 < K)
      __builtin_prefetch(&W[(size_t)(k0 + 32) * N + n0 + lane * 4], 0, 1);
    ASYNC_WAIT();
    __syncthreads();
    bf16x16 af[2], bfv[2];
#pragma unroll
    for (int i = 0; i < 2; i++) {
      const unsigned short* p = &lA[(wm + 16 * i + m) * 32 + 8 * hi];
      af[i] = frag2(p, p + 16);
    }
#pragma unroll
    for (int j = 0; j < 2; j++) {
      const unsigned short* p = &lB[(wn + 16 * j + m) * 32 + 8 * hi];
      bfv[j] = frag2(p, p + 16);
    }
#pragma unroll
    for (int i = 0; i < 2; i++)
#pragma unroll
      for (int j = 0; j < 2; j++) acc[i][j] = wmma_bf16(af[i], bfv[j], acc[i][j]);
    __syncthreads();
  }
#pragma unroll
  for (int i = 0; i < 2; i++)
#pragma unroll
    for (int j = 0; j < 2; j++)
#pragma unroll
      for (int r = 0; r < 8; r++) {
        size_t row = (size_t)(m0 + wm + 16 * i + r + 8 * hi);
        size_t col = (size_t)(n0 + wn + 16 * j + m);
        float v = acc[i][j][r];
        if (resid) v += resid[row * N + col];
        C[row * N + col] = v;
      }
}

// -------------------------------------------------- q/k rmsnorm + RoPE + v
__global__ __launch_bounds__(256) void k_rope(const float* __restrict__ Q,
                                              const float* __restrict__ K,
                                              const float* __restrict__ V,
                                              const float* __restrict__ qw,
                                              const float* __restrict__ kw,
                                              unsigned short* __restrict__ Qb,
                                              unsigned short* __restrict__ Kb,
                                              unsigned short* __restrict__ Vb) {
  int tok = blockIdx.x;
  int pos = tok & (S_ - 1);
  int lane = threadIdx.x & 31, wave = threadIdx.x >> 5;
  int i = lane;  // rotary pair index 0..31
  float inv = __powf(10000.0f, -(float)i * (1.0f / 32.0f));
  float sn, cs;
  __sincosf((float)pos * inv, &sn, &cs);
  for (int slot = wave; slot < 24; slot += 8) {
    if (slot < 16) {  // q heads
      const float* row = Q + (size_t)tok * (H_ * HD_) + slot * HD_;
      unsigned short* orow = Qb + (size_t)tok * (H_ * HD_) + slot * HD_;
      float x1 = row[2 * i], x2 = row[2 * i + 1];
      float ss = x1 * x1 + x2 * x2;
#pragma unroll
      for (int dd = 1; dd < 32; dd <<= 1) ss += __shfl_xor(ss, dd, 32);
      float rs = rsqrtf(ss * (1.0f / 64.0f) + 1e-5f);
      x1 *= rs * qw[2 * i]; x2 *= rs * qw[2 * i + 1];
      orow[2 * i]     = f2bf(x1 * cs - x2 * sn);
      orow[2 * i + 1] = f2bf(x1 * sn + x2 * cs);
    } else if (slot < 20) {  // k heads
      int h2 = slot - 16;
      const float* row = K + (size_t)tok * (KH_ * HD_) + h2 * HD_;
      unsigned short* orow = Kb + (size_t)tok * (KH_ * HD_) + h2 * HD_;
      float x1 = row[2 * i], x2 = row[2 * i + 1];
      float ss = x1 * x1 + x2 * x2;
#pragma unroll
      for (int dd = 1; dd < 32; dd <<= 1) ss += __shfl_xor(ss, dd, 32);
      float rs = rsqrtf(ss * (1.0f / 64.0f) + 1e-5f);
      x1 *= rs * kw[2 * i]; x2 *= rs * kw[2 * i + 1];
      orow[2 * i]     = f2bf(x1 * cs - x2 * sn);
      orow[2 * i + 1] = f2bf(x1 * sn + x2 * cs);
    } else {  // v heads: bf16 convert only
      int h2 = slot - 20;
      const float* row = V + (size_t)tok * (KH_ * HD_) + h2 * HD_;
      unsigned short* orow = Vb + (size_t)tok * (KH_ * HD_) + h2 * HD_;
      orow[2 * i]     = f2bf(row[2 * i]);
      orow[2 * i + 1] = f2bf(row[2 * i + 1]);
    }
  }
}

// ------------------------------------------------ flash attention (causal)
// grid.x = B * H * (S/128); 8 waves, each wave = one 16-row q tile.
__global__ __launch_bounds__(256) void k_attn(const unsigned short* __restrict__ Qb,
                                              const unsigned short* __restrict__ Kb,
                                              const unsigned short* __restrict__ Vb,
                                              float* __restrict__ Attn) {
  int bid = blockIdx.x;
  int qblk = bid & 15;
  int h = (bid >> 4) & 15;
  int b = bid >> 8;
  int kh = h >> 2;
  int tid = threadIdx.x, lane = tid & 31, wave = tid >> 5;
  int m = lane & 15, hi = lane >> 4;
  int q0 = qblk * 128 + wave * 16;
  size_t tb = (size_t)b * S_;
  __shared__ __attribute__((aligned(16))) unsigned short lK[32 * 64];
  __shared__ __attribute__((aligned(16))) unsigned short lV[32 * 64];
  __shared__ __attribute__((aligned(16))) unsigned short lP[8][16 * 32];

  bf16x16 qfr[2];
  {
    const unsigned short* qrow = Qb + (tb + q0 + m) * (size_t)(H_ * HD_) + h * HD_;
#pragma unroll
    for (int t = 0; t < 2; t++) {
      const unsigned short* p = qrow + 32 * t + 8 * hi;
      qfr[t] = frag2(p, p + 16);
    }
  }
  f32x8 o[4];
#pragma unroll
  for (int cf = 0; cf < 4; cf++) o[cf] = (f32x8)0.0f;
  float mrow[8], lrow[8];
#pragma unroll
  for (int r = 0; r < 8; r++) { mrow[r] = -1e30f; lrow[r] = 0.f; }

  int qmax = q0 + 15;
  int nkb = qblk * 4 + 4;
  for (int kb2 = 0; kb2 < nkb; kb2++) {
    {
      int r = tid >> 3, c = (tid & 7) * 8;
      size_t g = (tb + kb2 * 32 + r) * (size_t)(KH_ * HD_) + kh * HD_ + c;
      ASYNC_CP_B128(&Kb[g], &lK[r * 64 + c]);
      ASYNC_CP_B128(&Vb[g], &lV[r * 64 + c]);
    }
    ASYNC_WAIT();
    __syncthreads();
    if (kb2 * 32 <= qmax) {
      f32x8 s[2];
      s[0] = (f32x8)0.0f; s[1] = (f32x8)0.0f;
#pragma unroll
      for (int t = 0; t < 2; t++)
#pragma unroll
        for (int c = 0; c < 2; c++) {
          const unsigned short* p = &lK[(c * 16 + m) * 64 + 32 * t + 8 * hi];
          s[c] = wmma_bf16(qfr[t], frag2(p, p + 16), s[c]);
        }
#pragma unroll
      for (int c = 0; c < 2; c++) {
        int key = kb2 * 32 + c * 16 + m;
#pragma unroll
        for (int r = 0; r < 8; r++) {
          float v = s[c][r] * 0.125f;  // 1/sqrt(64)
          int qi = q0 + r + 8 * hi;
          s[c][r] = (key <= qi) ? v : -1e30f;
        }
      }
#pragma unroll
      for (int r = 0; r < 8; r++) {
        float mx = fmaxf(s[0][r], s[1][r]);
#pragma unroll
        for (int dd = 1; dd < 16; dd <<= 1) mx = fmaxf(mx, __shfl_xor(mx, dd, 32));
        float mnew = fmaxf(mrow[r], mx);
        float corr = __expf(mrow[r] - mnew);
        mrow[r] = mnew;
        float p0 = __expf(s[0][r] - mnew);
        float p1 = __expf(s[1][r] - mnew);
        s[0][r] = p0; s[1][r] = p1;
        float ps = p0 + p1;
#pragma unroll
        for (int dd = 1; dd < 16; dd <<= 1) ps += __shfl_xor(ps, dd, 32);
        lrow[r] = lrow[r] * corr + ps;
#pragma unroll
        for (int cf = 0; cf < 4; cf++) o[cf][r] *= corr;
      }
      // P back through LDS to switch C-layout -> A-layout
#pragma unroll
      for (int c = 0; c < 2; c++)
#pragma unroll
        for (int r = 0; r < 8; r++)
          lP[wave][(r + 8 * hi) * 32 + c * 16 + m] = f2bf(s[c][r]);
      const unsigned short* pp = &lP[wave][m * 32 + 16 * hi];
      bf16x16 pfr = frag2(pp, pp + 8);
#pragma unroll
      for (int cf = 0; cf < 4; cf++) {
        u16x16 vv;
#pragma unroll
        for (int e2 = 0; e2 < 16; e2++)
          vv[e2] = lV[(16 * hi + e2) * 64 + cf * 16 + m];
        o[cf] = wmma_bf16(pfr, __builtin_bit_cast(bf16x16, vv), o[cf]);
      }
    }
    __syncthreads();
  }
#pragma unroll
  for (int cf = 0; cf < 4; cf++)
#pragma unroll
    for (int r = 0; r < 8; r++) {
      size_t row = tb + q0 + r + 8 * hi;
      Attn[row * (size_t)(H_ * HD_) + h * HD_ + cf * 16 + m] = o[cf][r] / lrow[r];
    }
}

// -------------------------------------------------------------- gate mul
__global__ __launch_bounds__(256) void k_gate(const float* __restrict__ attn,
                                              const float* __restrict__ gpre,
                                              unsigned short* __restrict__ ao) {
  size_t idx = (size_t)blockIdx.x * 256 + threadIdx.x;
  float g = 1.f / (1.f + __expf(-gpre[idx]));
  ao[idx] = f2bf(attn[idx] * g);
}

// ------------------------------------------------------------- MoE router
__global__ __launch_bounds__(256) void k_router(const unsigned short* __restrict__ Xn,
                                                const float* __restrict__ RW,
                                                float* __restrict__ cw,
                                                int* __restrict__ cnt,
                                                float* __restrict__ probsum,
                                                int* __restrict__ idxl) {
  int tok = blockIdx.x * 8 + (threadIdx.x >> 5);
  int lane = threadIdx.x & 31;
  float acc[E_];
#pragma unroll
  for (int e = 0; e < E_; e++) acc[e] = 0.f;
  const unsigned short* xr = Xn + (size_t)tok * D_;
  for (int d = lane; d < D_; d += 32) {
    float xv = bf2f(xr[d]);
#pragma unroll
    for (int e = 0; e < E_; e++) acc[e] += xv * RW[d * E_ + e];
  }
#pragma unroll
  for (int e = 0; e < E_; e++)
#pragma unroll
    for (int dd = 1; dd < 32; dd <<= 1) acc[e] += __shfl_xor(acc[e], dd, 32);
  if (lane == 0) {
    float mx = acc[0];
#pragma unroll
    for (int e = 1; e < E_; e++) mx = fmaxf(mx, acc[e]);
    float p[E_], sum = 0.f;
#pragma unroll
    for (int e = 0; e < E_; e++) { p[e] = __expf(acc[e] - mx); sum += p[e]; }
#pragma unroll
    for (int e = 0; e < E_; e++) p[e] /= sum;
    int e1 = 0;
#pragma unroll
    for (int e = 1; e < E_; e++) if (p[e] > p[e1]) e1 = e;
    int e2 = (e1 == 0) ? 1 : 0;
#pragma unroll
    for (int e = 0; e < E_; e++) if (e != e1 && p[e] > p[e2]) e2 = e;
    float s2 = p[e1] + p[e2];
#pragma unroll
    for (int e = 0; e < E_; e++) cw[(size_t)tok * E_ + e] = 0.f;
    cw[(size_t)tok * E_ + e1] = p[e1] / s2;
    cw[(size_t)tok * E_ + e2] = p[e2] / s2;
#pragma unroll
    for (int e = 0; e < E_; e++)
      __hip_atomic_fetch_add(&probsum[e], p[e], __ATOMIC_RELAXED, __HIP_MEMORY_SCOPE_AGENT);
    int pos1 = atomicAdd(&cnt[e1], 1); idxl[e1 * T_ + pos1] = tok;
    int pos2 = atomicAdd(&cnt[e2], 1); idxl[e2 * T_ + pos2] = tok;
  }
}

__global__ void k_finalize(const int* cnt, const float* probsum, int* offs, float* auxp) {
  if (threadIdx.x == 0) {
    int o = 0; float aux = 0.f;
    for (int e = 0; e < E_; e++) { offs[e] = o; o += cnt[e]; }
    for (int e = 0; e < E_; e++) {
      float frac = (float)cnt[e] / (float)(T_ * 2);
      float pm = probsum[e] / (float)T_;
      aux += frac * pm;
    }
    auxp[0] = 0.01f * 8.f * aux;
  }
}

// -------------------------------- MoE pass 1: h = silu(tok@w1) * (tok@w3)
__global__ __launch_bounds__(256) void k_ffn1(const unsigned short* __restrict__ Xn,
                                              const float* __restrict__ W1,
                                              const float* __restrict__ W3,
                                              const int* __restrict__ idxl,
                                              const int* __restrict__ cnt,
                                              const int* __restrict__ offs,
                                              unsigned short* __restrict__ Hbuf) {
  int e = blockIdx.x >> 6;
  int m0 = (blockIdx.x & 63) * 64;
  int count = cnt[e];
  if (m0 >= count) return;
  int n0 = blockIdx.y * 128;
  const float* W1e = W1 + (size_t)e * D_ * FF_;
  const float* W3e = W3 + (size_t)e * D_ * FF_;
  __shared__ __attribute__((aligned(16))) unsigned short lA[64 * 32];
  __shared__ __attribute__((aligned(16))) unsigned short lB1[128 * 32];
  __shared__ __attribute__((aligned(16))) unsigned short lB3[128 * 32];
  __shared__ int rows[64];
  int tid = threadIdx.x, lane = tid & 31, wave = tid >> 5;
  int wm = (wave >> 2) * 32, wn = (wave & 3) * 32;
  int m = lane & 15, hi = lane >> 4;
  if (tid < 64) rows[tid] = (m0 + tid < count) ? idxl[e * T_ + m0 + tid] : -1;
  __syncthreads();
  f32x8 a1[2][2], a3[2][2];
#pragma unroll
  for (int i = 0; i < 2; i++)
#pragma unroll
    for (int j = 0; j < 2; j++) { a1[i][j] = (f32x8)0.0f; a3[i][j] = (f32x8)0.0f; }

  for (int k0 = 0; k0 < D_; k0 += 32) {
    {
      int r = tid >> 2, c = (tid & 3) * 8;
      int t = rows[r];
      if (t >= 0) {
        ASYNC_CP_B128(&Xn[(size_t)t * D_ + k0 + c], &lA[r * 32 + c]);
      } else {
        *(u16x8*)&lA[r * 32 + c] = (u16x8)(unsigned short)0;
      }
    }
    stage_w(W1e, FF_, k0, n0, lB1, lane, wave);
    stage_w(W3e, FF_, k0, n0, lB3, lane, wave);
    if (k0 + 32 < D_) {
      __builtin_prefetch(&W1e[(size_t)(k0 + 32) * FF_ + n0 + lane * 4], 0, 1);
      __builtin_prefetch(&W3e[(size_t)(k0 + 32) * FF_ + n0 + lane * 4], 0, 1);
    }
    ASYNC_WAIT();
    __syncthreads();
    bf16x16 af[2], b1[2], b3[2];
#pragma unroll
    for (int i = 0; i < 2; i++) {
      const unsigned short* p = &lA[(wm + 16 * i + m) * 32 + 8 * hi];
      af[i] = frag2(p, p + 16);
    }
#pragma unroll
    for (int j = 0; j < 2; j++) {
      const unsigned short* p1 = &lB1[(wn + 16 * j + m) * 32 + 8 * hi];
      const unsigned short* p3 = &lB3[(wn + 16 * j + m) * 32 + 8 * hi];
      b1[j] = frag2(p1, p1 + 16);
      b3[j] = frag2(p3, p3 + 16);
    }
#pragma unroll
    for (int i = 0; i < 2; i++)
#pragma unroll
      for (int j = 0; j < 2; j++) {
        a1[i][j] = wmma_bf16(af[i], b1[j], a1[i][j]);
        a3[i][j] = wmma_bf16(af[i], b3[j], a3[i][j]);
      }
    __syncthreads();
  }
#pragma unroll
  for (int i = 0; i < 2; i++)
#pragma unroll
    for (int j = 0; j < 2; j++)
#pragma unroll
      for (int r = 0; r < 8; r++) {
        int lr = wm + 16 * i + r + 8 * hi;
        if (m0 + lr < count) {
          float u = a1[i][j][r];
          float hval = (u / (1.f + __expf(-u))) * a3[i][j][r];
          Hbuf[(size_t)(offs[e] + m0 + lr) * FF_ + n0 + wn + 16 * j + m] = f2bf(hval);
        }
      }
}

// -------------------- MoE pass 2: out[tok] += cw[tok,e] * (h @ w2[e])
__global__ __launch_bounds__(256) void k_ffn2(const unsigned short* __restrict__ Hbuf,
                                              const float* __restrict__ W2,
                                              const int* __restrict__ idxl,
                                              const int* __restrict__ cnt,
                                              const int* __restrict__ offs,
                                              const float* __restrict__ cw,
                                              float* __restrict__ Out) {
  int e = blockIdx.x >> 6;
  int m0 = (blockIdx.x & 63) * 64;
  int count = cnt[e];
  if (m0 >= count) return;
  int n0 = blockIdx.y * 128;
  const float* W2e = W2 + (size_t)e * FF_ * D_;
  __shared__ __attribute__((aligned(16))) unsigned short lA[64 * 32];
  __shared__ __attribute__((aligned(16))) unsigned short lB[128 * 32];
  __shared__ int rows[64];
  int tid = threadIdx.x, lane = tid & 31, wave = tid >> 5;
  int wm = (wave >> 2) * 32, wn = (wave & 3) * 32;
  int m = lane & 15, hi = lane >> 4;
  if (tid < 64) rows[tid] = (m0 + tid < count) ? idxl[e * T_ + m0 + tid] : -1;
  __syncthreads();
  f32x8 acc[2][2];
#pragma unroll
  for (int i = 0; i < 2; i++)
#pragma unroll
    for (int j = 0; j < 2; j++) acc[i][j] = (f32x8)0.0f;

  int hbase = offs[e] + m0;
  for (int k0 = 0; k0 < FF_; k0 += 32) {
    {
      int r = tid >> 2, c = (tid & 3) * 8;
      if (m0 + r < count) {
        ASYNC_CP_B128(&Hbuf[(size_t)(hbase + r) * FF_ + k0 + c], &lA[r * 32 + c]);
      } else {
        *(u16x8*)&lA[r * 32 + c] = (u16x8)(unsigned short)0;
      }
    }
    stage_w(W2e, D_, k0, n0, lB, lane, wave);
    if (k0 + 32 < FF_)
      __builtin_prefetch(&W2e[(size_t)(k0 + 32) * D_ + n0 + lane * 4], 0, 1);
    ASYNC_WAIT();
    __syncthreads();
    bf16x16 af[2], bfv[2];
#pragma unroll
    for (int i = 0; i < 2; i++) {
      const unsigned short* p = &lA[(wm + 16 * i + m) * 32 + 8 * hi];
      af[i] = frag2(p, p + 16);
    }
#pragma unroll
    for (int j = 0; j < 2; j++) {
      const unsigned short* p = &lB[(wn + 16 * j + m) * 32 + 8 * hi];
      bfv[j] = frag2(p, p + 16);
    }
#pragma unroll
    for (int i = 0; i < 2; i++)
#pragma unroll
      for (int j = 0; j < 2; j++) acc[i][j] = wmma_bf16(af[i], bfv[j], acc[i][j]);
    __syncthreads();
  }
#pragma unroll
  for (int i = 0; i < 2; i++)
#pragma unroll
    for (int j = 0; j < 2; j++)
#pragma unroll
      for (int r = 0; r < 8; r++) {
        int lr = wm + 16 * i + r + 8 * hi;
        if (m0 + lr < count) {
          int tok = rows[lr];
          float v = acc[i][j][r] * cw[(size_t)tok * E_ + e];
          __hip_atomic_fetch_add(&Out[(size_t)tok * D_ + n0 + wn + 16 * j + m], v,
                                 __ATOMIC_RELAXED, __HIP_MEMORY_SCOPE_AGENT);
        }
      }
}

// ---------------------------------------------------------------- launcher
extern "C" void kernel_launch(void* const* d_in, const int* in_sizes, int n_in,
                              void* d_out, int out_size, void* d_ws, size_t ws_size,
                              hipStream_t stream) {
  const float* x   = (const float*)d_in[0];
  const float* anw = (const float*)d_in[1];
  const float* fnw = (const float*)d_in[2];
  const float* qnw = (const float*)d_in[3];
  const float* knw = (const float*)d_in[4];
  const float* wq  = (const float*)d_in[5];
  const float* wk  = (const float*)d_in[6];
  const float* wv  = (const float*)d_in[7];
  const float* wo  = (const float*)d_in[8];
  const float* wg  = (const float*)d_in[9];
  const float* rw  = (const float*)d_in[10];
  const float* w1  = (const float*)d_in[11];
  const float* w3  = (const float*)d_in[12];
  const float* w2  = (const float*)d_in[13];
  float* out = (float*)d_out;

  char* ws = (char*)d_ws;
  const size_t MB = 1024 * 1024;
  unsigned short* xn   = (unsigned short*)(ws);             // 8MB (reused as xn2)
  float*          qf   = (float*)(ws + 8 * MB);             // 16MB (reused as attn)
  float*          kf   = (float*)(ws + 24 * MB);            // 4MB
  float*          vf   = (float*)(ws + 28 * MB);            // 4MB
  unsigned short* ao   = (unsigned short*)(ws + 24 * MB);   // 8MB (aliases kf/vf after rope)
  float*          gpre = (float*)(ws + 32 * MB);            // 16MB
  unsigned short* qb   = (unsigned short*)(ws + 48 * MB);   // 8MB
  unsigned short* kb   = (unsigned short*)(ws + 56 * MB);   // 2MB
  unsigned short* vb   = (unsigned short*)(ws + 58 * MB);   // 2MB
  float*          cw   = (float*)(ws + 60 * MB);            // 128KB
  int*   cnt     = (int*)(ws + 60 * MB + 256 * 1024);
  float* probsum = (float*)(ws + 60 * MB + 256 * 1024 + 64);
  int*   offs    = (int*)(ws + 60 * MB + 256 * 1024 + 128);
  int*   idxl    = (int*)(ws + 61 * MB);                    // 128KB
  unsigned short* hbuf = (unsigned short*)(ws + 62 * MB);   // 32MB
  float* attn = qf;

  k_zero<<<1, 32, 0, stream>>>(cnt, probsum);
  k_rmsnorm<<<T_, 256, 0, stream>>>(x, anw, xn);
  k_gemm<<<dim3(T_ / 64, (H_ * HD_) / 128), 256, 0, stream>>>(xn, wq, qf, nullptr, H_ * HD_, D_);
  k_gemm<<<dim3(T_ / 64, (KH_ * HD_) / 128), 256, 0, stream>>>(xn, wk, kf, nullptr, KH_ * HD_, D_);
  k_gemm<<<dim3(T_ / 64, (KH_ * HD_) / 128), 256, 0, stream>>>(xn, wv, vf, nullptr, KH_ * HD_, D_);
  k_gemm<<<dim3(T_ / 64, (H_ * HD_) / 128), 256, 0, stream>>>(xn, wg, gpre, nullptr, H_ * HD_, D_);
  k_rope<<<T_, 256, 0, stream>>>(qf, kf, vf, qnw, knw, qb, kb, vb);
  k_attn<<<B_ * H_ * (S_ / 128), 256, 0, stream>>>(qb, kb, vb, attn);
  k_gate<<<(T_ * D_) / 256, 256, 0, stream>>>(attn, gpre, ao);
  k_gemm<<<dim3(T_ / 64, D_ / 128), 256, 0, stream>>>(ao, wo, out, x, D_, H_ * HD_);
  k_rmsnorm<<<T_, 256, 0, stream>>>(out, fnw, xn);
  k_router<<<T_ / 8, 256, 0, stream>>>(xn, rw, cw, cnt, probsum, idxl);
  k_finalize<<<1, 1, 0, stream>>>(cnt, probsum, offs, out + (size_t)T_ * D_);
  k_ffn1<<<dim3(E_ * 64, FF_ / 128), 256, 0, stream>>>(xn, w1, w3, idxl, cnt, offs, hbuf);
  k_ffn2<<<dim3(E_ * 64, D_ / 128), 256, 0, stream>>>(hbuf, w2, idxl, cnt, offs, cw, out);
  (void)in_sizes; (void)n_in; (void)out_size; (void)ws_size;
}
